// GATLayer_32684701123149
// MI455X (gfx1250) — matrix operand order repaired and checked by reading, not compiled, not run on hardware
//
#include <hip/hip_runtime.h>
#include <hip/hip_bf16.h>

// ---------------------------------------------------------------------------
// GAT layer, MI455X (gfx1250, wave32, WMMA).
//   N=8192, F_IN=512, F_OUT=256, heads=4
// Pipeline:
//   K1: Wh = x@W           (bf16 WMMA, f32 accum) -> Wh_f32 + Wh_bf16
//   K2: s_src/s_dst GEMVs  (wave32 shfl reduction)
//   K3: per-row softmax stats (max, 1/sum) streaming adj
//   K4: fused  h = softmax(mask(e))@Wh  (bf16 WMMA) + head-mean
// LDS operands are staged pre-swizzled into WMMA fragment order so fragment
// loads are 2x ds_load_b128 per v16bf (no scalar u16 gathers / b16 packing).
// ---------------------------------------------------------------------------

typedef __bf16 bf16_t;
typedef __attribute__((ext_vector_type(16))) __bf16 v16bf;
typedef __attribute__((ext_vector_type(8)))  float  v8f;

#define NN    8192
#define FIN   512
#define FOUT  256
#define FH    64      // FOUT / heads

__device__ inline unsigned int f32_to_bf16_rne(float f) {
    unsigned int u = __builtin_bit_cast(unsigned int, f);
    unsigned int lsb = (u >> 16) & 1u;
    u += 0x7fffu + lsb;                    // round to nearest even
    return u >> 16;
}
__device__ inline float lrelu(float v) { return v > 0.f ? v : 0.2f * v; }

// A-matrix (16x32 bf16) fragment slot for (row 0..15, K 0..31):
//   lane = row + 16*((K>>3)&1);  element = (K&7) + 8*(K>>4)
// B-matrix (32x16 bf16) fragment slot for (K 0..31, col 0..15):
//   lane = col + 16*((K>>3)&1); element = (K&7) + 8*(K>>4)
// Fragment storage: [tile][lane][8 x u32]  (32 bytes per lane, b128-aligned)

// ---------------------------------------------------------------------------
// K1: Wh = x @ W.  Block = 32 rows x 256 cols; 8 waves x (2 m-tiles x 2 n-tiles).
// ---------------------------------------------------------------------------
__global__ __launch_bounds__(256) void k_gemm_xw(
    const float* __restrict__ x, const float* __restrict__ W,
    float* __restrict__ Wh, unsigned short* __restrict__ Whb)
{
    __shared__ unsigned int sB32[16 * 32 * 8];   // W fragments, 16 n-tiles (16 KB)
    __shared__ unsigned int sA32[2 * 32 * 8];    // x fragments, 2 m-tiles  (2 KB)

    const int tid  = threadIdx.x;
    const int lane = tid & 31;
    const int wave = tid >> 5;
    const int row0 = blockIdx.x * 32;

    v8f acc[2][2] = {};

    for (int k0 = 0; k0 < FIN; k0 += 32) {
        // stage W chunk [k0..k0+31][0..255] -> bf16 pairs in fragment order
        for (int idx = tid; idx < 4096; idx += 256) {
            int cc = idx & 255;
            int kp = idx >> 8;                 // K pair index, K = 2*kp
            int K  = kp * 2;
            unsigned int b0 = f32_to_bf16_rne(W[(k0 + K)     * FOUT + cc]);
            unsigned int b1 = f32_to_bf16_rne(W[(k0 + K + 1) * FOUT + cc]);
            int nt = cc >> 4, col = cc & 15;
            int laneW = col + 16 * ((K >> 3) & 1);
            int e = (K & 7) + 8 * (K >> 4);
            sB32[(nt * 32 + laneW) * 8 + (e >> 1)] = b0 | (b1 << 16);
        }
        // stage x tile [row0..+31][k0..k0+31] -> fragment order
        for (int idx = tid; idx < 512; idx += 256) {
            int kp = idx & 15;
            int rr = idx >> 4;                 // 0..31
            int K  = kp * 2;
            const float* xp = x + (size_t)(row0 + rr) * FIN + k0 + K;
            unsigned int a0 = f32_to_bf16_rne(xp[0]);
            unsigned int a1 = f32_to_bf16_rne(xp[1]);
            int mt = rr >> 4, rloc = rr & 15;
            int laneW = rloc + 16 * ((K >> 3) & 1);
            int e = (K & 7) + 8 * (K >> 4);
            sA32[(mt * 32 + laneW) * 8 + (e >> 1)] = a0 | (a1 << 16);
        }
        __syncthreads();

        v16bf afrag0 = *(const v16bf*)&sA32[(0 * 32 + lane) * 8];
        v16bf afrag1 = *(const v16bf*)&sA32[(1 * 32 + lane) * 8];
        #pragma unroll
        for (int t = 0; t < 2; ++t) {
            int nt = wave * 2 + t;
            v16bf bfrag = *(const v16bf*)&sB32[(nt * 32 + lane) * 8];
            acc[0][t] = __builtin_amdgcn_wmma_f32_16x16x32_bf16(
                false, afrag0, false, bfrag, (short)0, acc[0][t], false, false);
            acc[1][t] = __builtin_amdgcn_wmma_f32_16x16x32_bf16(
                false, afrag1, false, bfrag, (short)0, acc[1][t], false, false);
        }
        __syncthreads();
    }

    // store: C/D layout — VGPR r: lanes 0-15 => M=r, lanes 16-31 => M=r+8
    #pragma unroll
    for (int mt = 0; mt < 2; ++mt) {
        #pragma unroll
        for (int t = 0; t < 2; ++t) {
            int c = (wave * 2 + t) * 16 + (lane & 15);
            int rbase = row0 + mt * 16 + ((lane < 16) ? 0 : 8);
            #pragma unroll
            for (int r = 0; r < 8; ++r) {
                float v = acc[mt][t][r];
                Wh [(size_t)(rbase + r) * FOUT + c] = v;
                Whb[(size_t)(rbase + r) * FOUT + c] = (unsigned short)f32_to_bf16_rne(v);
            }
        }
    }
}

// ---------------------------------------------------------------------------
// K2: s_src[i] = Wh[i,:].a[:256], s_dst[i] = Wh[i,:].a[256:].  Wave per row.
// ---------------------------------------------------------------------------
__global__ __launch_bounds__(256) void k_scores(
    const float* __restrict__ Wh, const float* __restrict__ a,
    float* __restrict__ s_src, float* __restrict__ s_dst)
{
    const int tid = threadIdx.x;
    const int lane = tid & 31, wave = tid >> 5;
    const int row = blockIdx.x * 8 + wave;

    float s1 = 0.f, s2 = 0.f;
    for (int k = lane; k < FOUT; k += 32) {
        float w = Wh[(size_t)row * FOUT + k];
        s1 += w * a[k];
        s2 += w * a[FOUT + k];
    }
    #pragma unroll
    for (int o = 16; o > 0; o >>= 1) {
        s1 += __shfl_xor(s1, o, 32);
        s2 += __shfl_xor(s2, o, 32);
    }
    if (lane == 0) { s_src[row] = s1; s_dst[row] = s2; }
}

// ---------------------------------------------------------------------------
// K3: per-row softmax stats. att[i,j] = adj? lrelu(s_src+s_dst) : 0, and the
// zeros DO participate. Online (max, sum-exp), LDS tree combine.
// ---------------------------------------------------------------------------
__global__ __launch_bounds__(256) void k_rowstats(
    const int* __restrict__ adj, const float* __restrict__ s_src,
    const float* __restrict__ s_dst, float* __restrict__ rowmax,
    float* __restrict__ rowinv)
{
    __shared__ float sm[256];
    __shared__ float sl[256];

    const int row = blockIdx.x;
    const int tid = threadIdx.x;
    const float ss = s_src[row];
    const int* __restrict__ arow = adj + (size_t)row * NN;

    float m = -__builtin_inff();
    float l = 0.f;
    for (int j = tid; j < NN; j += 256) {
        __builtin_prefetch(arow + j + 2048, 0, 1);   // global_prefetch_b8
        int av = arow[j];
        float e = av ? lrelu(ss + s_dst[j]) : 0.f;
        if (e > m) { l = l * __expf(m - e) + 1.f; m = e; }
        else       { l += __expf(e - m); }
    }
    sm[tid] = m; sl[tid] = l;
    __syncthreads();
    for (int s = 128; s > 0; s >>= 1) {
        if (tid < s) {
            float m2 = sm[tid + s], l2 = sl[tid + s];
            float mm = fmaxf(sm[tid], m2);
            sl[tid] = sl[tid] * __expf(sm[tid] - mm) + l2 * __expf(m2 - mm);
            sm[tid] = mm;
        }
        __syncthreads();
    }
    if (tid == 0) {
        rowmax[row] = sm[0];
        rowinv[row] = 1.f / sl[0];
    }
}

// ---------------------------------------------------------------------------
// K4: fused h = P @ Wh (P regenerated on the fly from adj + stats) + head-mean.
// Block = 32 rows x 256 cols; 8 waves x (2 m-tiles x 2 n-tiles); K-loop step 32.
// ---------------------------------------------------------------------------
__global__ __launch_bounds__(256) void k_attn_gemm(
    const int* __restrict__ adj, const unsigned short* __restrict__ Whb,
    const float* __restrict__ s_src, const float* __restrict__ s_dst,
    const float* __restrict__ rowmax, const float* __restrict__ rowinv,
    float* __restrict__ out)
{
    __shared__ unsigned int sB32[16 * 32 * 8];   // Wh bf16 fragments (16 KB)
    __shared__ unsigned int sP32[2 * 32 * 8];    // probability fragments (2 KB)
    __shared__ float sH[32 * 260];               // 32x256 f32 for head-mean
    __shared__ float sSrc[32], sMax[32], sInv[32];

    const int tid  = threadIdx.x;
    const int lane = tid & 31;
    const int wave = tid >> 5;
    const int row0 = blockIdx.x * 32;

    if (tid < 32) {
        sSrc[tid] = s_src[row0 + tid];
        sMax[tid] = rowmax[row0 + tid];
        sInv[tid] = rowinv[row0 + tid];
    }
    __syncthreads();

    v8f acc[2][2] = {};

    for (int j0 = 0; j0 < NN; j0 += 32) {
        // regenerate probability tile 32x32: p = exp(att - m) * inv, bf16 pairs
        for (int idx = tid; idx < 512; idx += 256) {
            int kp = idx & 15;                 // K pair
            int rr = idx >> 4;                 // 0..31
            int K  = kp * 2;
            int r  = row0 + rr;
            int j  = j0 + K;
            const int* ap = adj + (size_t)r * NN + j;
            __builtin_prefetch(ap + 32, 0, 1);
            int a0 = ap[0], a1 = ap[1];
            float ssr = sSrc[rr], mI = sMax[rr], iv = sInv[rr];
            float e0 = a0 ? lrelu(ssr + s_dst[j])     : 0.f;
            float e1 = a1 ? lrelu(ssr + s_dst[j + 1]) : 0.f;
            unsigned int p0 = f32_to_bf16_rne(__expf(e0 - mI) * iv);
            unsigned int p1 = f32_to_bf16_rne(__expf(e1 - mI) * iv);
            int mt = rr >> 4, rloc = rr & 15;
            int laneW = rloc + 16 * ((K >> 3) & 1);
            int e = (K & 7) + 8 * (K >> 4);
            sP32[(mt * 32 + laneW) * 8 + (e >> 1)] = p0 | (p1 << 16);
        }
        // stage Wh bf16 chunk [j0..j0+31][0..255] in fragment order (L2-resident)
        for (int idx = tid; idx < 4096; idx += 256) {
            int cc = idx & 255;
            int kp = idx >> 8;
            int K  = kp * 2;
            unsigned int b0 = Whb[(size_t)(j0 + K)     * FOUT + cc];
            unsigned int b1 = Whb[(size_t)(j0 + K + 1) * FOUT + cc];
            int nt = cc >> 4, col = cc & 15;
            int laneW = col + 16 * ((K >> 3) & 1);
            int e = (K & 7) + 8 * (K >> 4);
            sB32[(nt * 32 + laneW) * 8 + (e >> 1)] = b0 | (b1 << 16);
        }
        __syncthreads();

        v16bf afrag0 = *(const v16bf*)&sP32[(0 * 32 + lane) * 8];
        v16bf afrag1 = *(const v16bf*)&sP32[(1 * 32 + lane) * 8];
        #pragma unroll
        for (int t = 0; t < 2; ++t) {
            int nt = wave * 2 + t;
            v16bf bfrag = *(const v16bf*)&sB32[(nt * 32 + lane) * 8];
            acc[0][t] = __builtin_amdgcn_wmma_f32_16x16x32_bf16(
                false, afrag0, false, bfrag, (short)0, acc[0][t], false, false);
            acc[1][t] = __builtin_amdgcn_wmma_f32_16x16x32_bf16(
                false, afrag1, false, bfrag, (short)0, acc[1][t], false, false);
        }
        __syncthreads();
    }

    // spill accumulators to LDS, then head-mean: out[r][c] over 4 heads of 64
    #pragma unroll
    for (int mt = 0; mt < 2; ++mt) {
        #pragma unroll
        for (int t = 0; t < 2; ++t) {
            int c = (wave * 2 + t) * 16 + (lane & 15);
            int rbase = mt * 16 + ((lane < 16) ? 0 : 8);
            #pragma unroll
            for (int r = 0; r < 8; ++r)
                sH[(rbase + r) * 260 + c] = acc[mt][t][r];
        }
    }
    __syncthreads();
    for (int i = tid; i < 32 * FH; i += 256) {
        int rr = i >> 6, cc = i & 63;
        const float* hr = &sH[rr * 260];
        float v = 0.25f * (hr[cc] + hr[64 + cc] + hr[128 + cc] + hr[192 + cc]);
        out[(size_t)(row0 + rr) * FH + cc] = v;
    }
}

// ---------------------------------------------------------------------------
extern "C" void kernel_launch(void* const* d_in, const int* in_sizes, int n_in,
                              void* d_out, int out_size, void* d_ws, size_t ws_size,
                              hipStream_t stream) {
    const float* x   = (const float*)d_in[0];   // [8192, 512]
    const int*   adj = (const int*)d_in[1];     // [8192, 8192]
    const float* W   = (const float*)d_in[2];   // [512, 256]
    const float* a   = (const float*)d_in[3];   // [512, 1]
    // d_in[4] = heads (4), hardcoded.

    char* ws = (char*)d_ws;
    float*          Wh     = (float*)ws;                                     // 8 MiB
    unsigned short* Whb    = (unsigned short*)(ws + (size_t)NN * FOUT * 4);  // 4 MiB
    float*          s_src  = (float*)(ws + (size_t)NN * FOUT * 6);
    float*          s_dst  = s_src + NN;
    float*          rmax   = s_dst + NN;
    float*          rinv   = rmax + NN;

    k_gemm_xw  <<<NN / 32, 256, 0, stream>>>(x, W, Wh, Whb);
    k_scores   <<<NN / 8,  256, 0, stream>>>(Wh, a, s_src, s_dst);
    k_rowstats <<<NN,      256, 0, stream>>>(adj, s_src, s_dst, rmax, rinv);
    k_attn_gemm<<<NN / 32, 256, 0, stream>>>(adj, Whb, s_src, s_dst, rmax, rinv,
                                             (float*)d_out);
}
